// SwinBlock_68238440399252
// MI455X (gfx1250) — matrix operand order repaired and checked by reading, not compiled, not run on hardware
//
#include <hip/hip_runtime.h>
#include <hip/hip_bf16.h>

typedef __attribute__((ext_vector_type(16))) _Float16 v16h;
typedef __attribute__((ext_vector_type(8)))  _Float16 v8h;
typedef __attribute__((ext_vector_type(8)))  float    v8f;

#define WSZ    7
#define SHIFT_ 3
#define NH_    12
#define DIM_   384
#define HID_   1536
#define NTOK   49            // tokens per 7x7 window
#define NPAD   64            // padded to 4x 16-row WMMA tiles
#define HD_    32            // head dim
#define QKC    768           // q+k columns kept row-major in LDS

// ---------------- wide fragment loads (two b128 per fragment) ---------------

__device__ __forceinline__ v16h cat8(v8h lo, v8h hi) {
  return __builtin_shufflevector(lo, hi, 0, 1, 2, 3, 4, 5, 6, 7,
                                         8, 9, 10, 11, 12, 13, 14, 15);
}

// A-matrix 16x32 f16 (row-major MxK source): lane r=lane&15 holds row M=r;
// lanes 0-15: K 0..7 & 16..23, lanes 16-31: K 8..15 & 24..31 (ISA 7.12.2).
// Per lane: 8 contiguous halves at kb, 8 contiguous at kb+16 -> 2x b128.
__device__ __forceinline__ v16h load_afrag(const _Float16* base, int ld) {
  const int lane = threadIdx.x & 31;
  const _Float16* p = base + (lane & 15) * ld + ((lane >> 4) << 3);
  v8h lo = *(const v8h*)p;
  v8h hi = *(const v8h*)(p + 16);
  return cat8(lo, hi);
}

// B-matrix 32x16 f16 where the source is stored N-major (NxK row-major),
// i.e. element (k,n) = src[n*ld + k]. Lane n=lane&15 holds column N=n;
// lanes 0-15: K 0..7 & 16..23, lanes 16-31: K 8..15 & 24..31 (matches A).
__device__ __forceinline__ v16h load_bfrag_nmajor(const _Float16* base, int ld) {
  const int lane = threadIdx.x & 31;
  const _Float16* p = base + (lane & 15) * ld + ((lane >> 4) << 3);
  v8h lo = *(const v8h*)p;
  v8h hi = *(const v8h*)(p + 16);
  return cat8(lo, hi);
}

__device__ __forceinline__ v8f wmma_f16(v16h a, v16h b, v8f c) {
  // 8 args: (neg_a, A, neg_b, B, c_mod, C, reuse_a, reuse_b)
  return __builtin_amdgcn_wmma_f32_16x16x32_f16(false, a, false, b, (short)0, c,
                                                false, false);
}

// ------------- weight convert fp32 (KxN) -> f16 transposed (NxK) ------------
__global__ void cvt_transpose_f16(const float* __restrict__ s,
                                  _Float16* __restrict__ d, int K, int N) {
  int idx = blockIdx.x * 256 + threadIdx.x;
  if (idx < K * N) {
    int k = idx / N, n = idx - k * N;
    d[n * K + k] = (_Float16)s[idx];
  }
}

// ---------------- fused LN1 + shifted-window attention + proj ----------------
// 1 workgroup (128 thr = 4 wave32) per window. 2048 windows.
// LDS: S(4*64*64 f32)=64K | S16(4*64*64 f16)=32K | win/out(64*384 f16)=48K |
//      qk(64*768 f16)=96K | Vt(384*64 f16, V transposed)=48K  -> 288 KB.
__global__ __launch_bounds__(128) void swin_attn_kernel(
    const float* __restrict__ x, const float* __restrict__ attn_mask,
    const float* __restrict__ n1g, const float* __restrict__ n1b,
    const _Float16* __restrict__ qkvwT, const float* __restrict__ qkvb,
    const float* __restrict__ rpb, const _Float16* __restrict__ projwT,
    const float* __restrict__ projb, float* __restrict__ out)
{
  extern __shared__ __align__(16) char smem[];
  float*    S    = (float*)smem;                          // 65536 B
  _Float16* S16  = (_Float16*)(smem + 65536);             // 32768 B
  _Float16* winL = (_Float16*)(smem + 98304);             // 49152 B (64x384)
  _Float16* qkL  = (_Float16*)(smem + 147456);            // 98304 B (64x768)
  _Float16* Vt   = (_Float16*)(smem + 245760);            // 49152 B (384x64)
  _Float16* outL = winL;   // reused after QKV GEMM completes

  const int wid  = blockIdx.x;        // 0..2047
  const int bimg = wid >> 6;          // image
  const int w2   = wid & 63;          // window within image
  const int wr   = w2 >> 3;
  const int wc   = w2 & 7;
  const int lane = threadIdx.x & 31;
  const int wave = threadIdx.x >> 5;
  const float scale = 0.17677669529663689f;   // 32^-0.5

  // ---- LN1 + roll(-3,-3) + window partition gather -> winL ----
  for (int r = wave; r < NPAD; r += 4) {
    if (r < NTOK) {
      int rr = r / 7, cc = r - rr * 7;
      int h0 = (wr * 7 + rr + SHIFT_) % 56;
      int w0 = (wc * 7 + cc + SHIFT_) % 56;
      const float* xp = x + ((size_t)bimg * 3136 + h0 * 56 + w0) * DIM_;
      float s = 0.f, s2 = 0.f;
      for (int c0 = lane; c0 < DIM_; c0 += 32) { float v = xp[c0]; s += v; s2 += v*v; }
#pragma unroll
      for (int o = 16; o > 0; o >>= 1) { s += __shfl_xor(s, o, 32); s2 += __shfl_xor(s2, o, 32); }
      float mean = s * (1.f / DIM_);
      float rstd = rsqrtf(s2 * (1.f / DIM_) - mean * mean + 1e-5f);
      for (int c0 = lane; c0 < DIM_; c0 += 32)
        winL[r*DIM_ + c0] = (_Float16)((xp[c0] - mean) * rstd * n1g[c0] + n1b[c0]);
    } else {
      for (int c0 = lane; c0 < DIM_; c0 += 32) winL[r*DIM_ + c0] = (_Float16)0.f;
    }
  }
  __syncthreads();

  // ---- QKV GEMM: (64x384)@(384x1152); q,k -> qkL row-major, v -> Vt^T ----
  for (int t = wave; t < 4*72; t += 4) {
    int tm = t & 3, tn = t >> 2;
    v8f c = {};
#pragma unroll 4
    for (int kk = 0; kk < 12; ++kk) {
      v16h a = load_afrag(winL + tm*16*DIM_ + kk*32, DIM_);
      v16h b = load_bfrag_nmajor(qkvwT + (tn*16)*DIM_ + kk*32, DIM_);
      c = wmma_f16(a, b, c);
    }
    int rr  = (lane >> 4) << 3;
    int cl  = lane & 15;
    int col = tn*16 + cl;
    float bias = qkvb[col];
    if (col < DIM_) {                 // q: scaled
#pragma unroll
      for (int j = 0; j < 8; ++j)
        qkL[(tm*16 + rr + j)*QKC + col] = (_Float16)((c[j] + bias) * scale);
    } else if (col < 2*DIM_) {        // k
#pragma unroll
      for (int j = 0; j < 8; ++j)
        qkL[(tm*16 + rr + j)*QKC + col] = (_Float16)(c[j] + bias);
    } else {                          // v: transposed store (row k, col = token)
      int vc = col - 2*DIM_;
#pragma unroll
      for (int j = 0; j < 8; ++j)
        Vt[vc*NPAD + (tm*16 + rr + j)] = (_Float16)(c[j] + bias);
    }
  }
  __syncthreads();

  // ---- per-head attention; each wave owns heads {w, w+4, w+8} ----
  float*    Sw   = S   + wave * 64*64;
  _Float16* S16w = S16 + wave * 64*64;
  const float* maskw = attn_mask + (size_t)w2 * NTOK * NTOK;
  for (int h = wave; h < NH_; h += 4) {
    const _Float16* Q  = qkL + h*HD_;
    const _Float16* K  = qkL + DIM_ + h*HD_;
    const _Float16* Vh = Vt + (h*HD_)*NPAD;       // N-major (col n, k contiguous)

    // S = Q K^T (64x64 padded); K-dim = 32 = exactly one WMMA per tile.
    // B = K^T: element (k,n) = K[n][k] -> N-major gather on row-major K.
#pragma unroll
    for (int tm = 0; tm < 4; ++tm)
#pragma unroll
      for (int tn = 0; tn < 4; ++tn) {
        v8f c = {};
        v16h a = load_afrag(Q + tm*16*QKC, QKC);
        v16h b = load_bfrag_nmajor(K + tn*16*QKC, QKC);
        c = wmma_f16(a, b, c);
        int rr = (lane >> 4) << 3;
        int cl = lane & 15;
#pragma unroll
        for (int j = 0; j < 8; ++j)
          Sw[(tm*16 + rr + j)*64 + tn*16 + cl] = c[j];
      }

    // softmax rows with rel-pos bias + shifted-window mask
#pragma unroll
    for (int rb = 0; rb < 64; rb += 32) {
      int r = rb + lane;
      if (r < NTOK) {
        int r1 = r / 7, c1 = r - r1*7;
        float m = -1e30f;
        for (int j2 = 0; j2 < NTOK; ++j2) {
          int r2 = j2 / 7, c2 = j2 - r2*7;
          int ridx = (r1 - r2 + 6) * 13 + (c1 - c2 + 6);
          float v = Sw[r*64 + j2] + rpb[ridx*NH_ + h] + maskw[r*NTOK + j2];
          Sw[r*64 + j2] = v;
          m = fmaxf(m, v);
        }
        float sum = 0.f;
        for (int j2 = 0; j2 < NTOK; ++j2) {
          float e = __expf(Sw[r*64 + j2] - m);
          Sw[r*64 + j2] = e; sum += e;
        }
        float inv = 1.f / sum;
        for (int j2 = 0; j2 < NTOK; ++j2) S16w[r*64 + j2] = (_Float16)(Sw[r*64+j2] * inv);
        for (int j2 = NTOK; j2 < 64; ++j2) S16w[r*64 + j2] = (_Float16)0.f;
      } else {
        for (int j2 = 0; j2 < 64; ++j2) S16w[r*64 + j2] = (_Float16)0.f;
      }
    }

    // O = softmax(S) @ V : (64x64)@(64x32); V read N-major from Vt
#pragma unroll
    for (int tm = 0; tm < 4; ++tm)
#pragma unroll
      for (int tn = 0; tn < 2; ++tn) {
        v8f c = {};
#pragma unroll
        for (int kk = 0; kk < 2; ++kk) {
          v16h a = load_afrag(S16w + tm*16*64 + kk*32, 64);
          v16h b = load_bfrag_nmajor(Vh + (tn*16)*NPAD + kk*32, NPAD);
          c = wmma_f16(a, b, c);
        }
        int rr = (lane >> 4) << 3;
        int cl = lane & 15;
#pragma unroll
        for (int j = 0; j < 8; ++j)
          outL[(tm*16 + rr + j)*DIM_ + h*HD_ + tn*16 + cl] = (_Float16)c[j];
      }
  }
  __syncthreads();

  // ---- proj GEMM (64x384)@(384x384) + residual + inverse-shift scatter ----
  for (int t = wave; t < 4*24; t += 4) {
    int tm = t & 3, tn = t >> 2;
    v8f c = {};
#pragma unroll 4
    for (int kk = 0; kk < 12; ++kk) {
      v16h a = load_afrag(outL + tm*16*DIM_ + kk*32, DIM_);
      v16h b = load_bfrag_nmajor(projwT + (tn*16)*DIM_ + kk*32, DIM_);
      c = wmma_f16(a, b, c);
    }
    int rr0 = (lane >> 4) << 3;
    int cl  = lane & 15;
    int col = tn*16 + cl;
    float pb = projb[col];
#pragma unroll
    for (int j = 0; j < 8; ++j) {
      int r = tm*16 + rr0 + j;
      if (r < NTOK) {
        int rw = r / 7, cw = r - rw*7;
        int h0 = (wr*7 + rw + SHIFT_) % 56;
        int w0 = (wc*7 + cw + SHIFT_) % 56;
        size_t tok = (size_t)bimg * 3136 + h0*56 + w0;
        out[tok*DIM_ + col] = x[tok*DIM_ + col] + c[j] + pb;
      }
    }
  }
}

// ---------------- fused LN2 + MLP (fc1 + GELU + fc2) + residual --------------
// 1 workgroup (256 thr = 8 wave32) per 64-row slab. 1568 blocks.
// LDS: xln(64*384 f16)=48K | h1(64*1536 f16)=192K -> 240 KB.
__global__ __launch_bounds__(256) void swin_mlp_kernel(
    const float* __restrict__ n2g, const float* __restrict__ n2b,
    const _Float16* __restrict__ fc1wT, const float* __restrict__ fc1b,
    const _Float16* __restrict__ fc2wT, const float* __restrict__ fc2b,
    float* __restrict__ xio)
{
  extern __shared__ __align__(16) char smem[];
  _Float16* xln = (_Float16*)smem;              // 64*384
  _Float16* h1  = xln + 64*DIM_;                // 64*1536
  const int lane = threadIdx.x & 31;
  const int wave = threadIdx.x >> 5;
  const size_t rowbase = (size_t)blockIdx.x * 64;

  for (int r = wave; r < 64; r += 8) {
    const float* xp = xio + (rowbase + r) * DIM_;
    float s = 0.f, s2 = 0.f;
    for (int c0 = lane; c0 < DIM_; c0 += 32) { float v = xp[c0]; s += v; s2 += v*v; }
#pragma unroll
    for (int o = 16; o > 0; o >>= 1) { s += __shfl_xor(s, o, 32); s2 += __shfl_xor(s2, o, 32); }
    float mean = s * (1.f / DIM_);
    float rstd = rsqrtf(s2 * (1.f / DIM_) - mean * mean + 1e-5f);
    for (int c0 = lane; c0 < DIM_; c0 += 32)
      xln[r*DIM_ + c0] = (_Float16)((xp[c0] - mean) * rstd * n2g[c0] + n2b[c0]);
  }
  __syncthreads();

  // fc1: (64x384)@(384x1536) + bias, exact GELU (erf)
  for (int t = wave; t < 4*96; t += 8) {
    int tm = t & 3, tn = t >> 2;
    v8f c = {};
#pragma unroll 4
    for (int kk = 0; kk < 12; ++kk) {
      v16h a = load_afrag(xln + tm*16*DIM_ + kk*32, DIM_);
      v16h b = load_bfrag_nmajor(fc1wT + (tn*16)*DIM_ + kk*32, DIM_);
      c = wmma_f16(a, b, c);
    }
    int rr = (lane >> 4) << 3, cl = lane & 15, col = tn*16 + cl;
    float bb = fc1b[col];
#pragma unroll
    for (int j = 0; j < 8; ++j) {
      float v = c[j] + bb;
      v = 0.5f * v * (1.f + erff(v * 0.70710678118654752f));
      h1[(tm*16 + rr + j)*HID_ + col] = (_Float16)v;
    }
  }
  __syncthreads();

  // fc2: (64x1536)@(1536x384) + bias, residual accumulate
  for (int t = wave; t < 4*24; t += 8) {
    int tm = t & 3, tn = t >> 2;
    v8f c = {};
#pragma unroll 4
    for (int kk = 0; kk < 48; ++kk) {
      v16h a = load_afrag(h1 + tm*16*HID_ + kk*32, HID_);
      v16h b = load_bfrag_nmajor(fc2wT + (tn*16)*HID_ + kk*32, HID_);
      c = wmma_f16(a, b, c);
    }
    int rr = (lane >> 4) << 3, cl = lane & 15, col = tn*16 + cl;
    float bb = fc2b[col];
#pragma unroll
    for (int j = 0; j < 8; ++j) {
      size_t tok = rowbase + tm*16 + rr + j;
      xio[tok*DIM_ + col] += c[j] + bb;
    }
  }
}

// ---------------- host launch ------------------------------------------------
extern "C" void kernel_launch(void* const* d_in, const int* in_sizes, int n_in,
                              void* d_out, int out_size, void* d_ws, size_t ws_size,
                              hipStream_t stream)
{
  const float* x     = (const float*)d_in[0];
  const float* amask = (const float*)d_in[1];
  const float* n1g   = (const float*)d_in[2];
  const float* n1b   = (const float*)d_in[3];
  const float* qkvw  = (const float*)d_in[4];
  const float* qkvb  = (const float*)d_in[5];
  const float* rpb   = (const float*)d_in[6];
  const float* projw = (const float*)d_in[7];
  const float* projb = (const float*)d_in[8];
  const float* n2g   = (const float*)d_in[9];
  const float* n2b   = (const float*)d_in[10];
  const float* fc1w  = (const float*)d_in[11];
  const float* fc1b  = (const float*)d_in[12];
  const float* fc2w  = (const float*)d_in[13];
  const float* fc2b  = (const float*)d_in[14];
  float* out = (float*)d_out;

  const int n_qkvw = DIM_ * 3 * DIM_, n_projw = DIM_ * DIM_;
  const int n_fc1w = DIM_ * HID_,     n_fc2w  = HID_ * DIM_;

  _Float16* qkvwT  = (_Float16*)d_ws;           // ~3.5 MB total in d_ws
  _Float16* projwT = qkvwT  + n_qkvw;
  _Float16* fc1wT  = projwT + n_projw;
  _Float16* fc2wT  = fc1wT  + n_fc1w;

  cvt_transpose_f16<<<(n_qkvw  + 255)/256, 256, 0, stream>>>(qkvw,  qkvwT,  DIM_, 3*DIM_);
  cvt_transpose_f16<<<(n_projw + 255)/256, 256, 0, stream>>>(projw, projwT, DIM_, DIM_);
  cvt_transpose_f16<<<(n_fc1w  + 255)/256, 256, 0, stream>>>(fc1w,  fc1wT,  DIM_, HID_);
  cvt_transpose_f16<<<(n_fc2w  + 255)/256, 256, 0, stream>>>(fc2w,  fc2wT,  HID_, DIM_);

  const size_t attn_lds = 65536 + 32768 + 49152 + 98304 + 49152;   // 294912 B
  swin_attn_kernel<<<2048, 128, attn_lds, stream>>>(
      x, amask, n1g, n1b, qkvwT, qkvb, rpb, projwT, projb, out);

  const size_t mlp_lds = (size_t)64*DIM_*2 + (size_t)64*HID_*2;    // 245760 B
  swin_mlp_kernel<<<100352/64, 256, mlp_lds, stream>>>(
      n2g, n2b, fc1wT, fc1b, fc2wT, fc2b, out);
}
